// HyperAttnBlock_78297253806433
// MI455X (gfx1250) — matrix operand order repaired and checked by reading, not compiled, not run on hardware
//
#include <hip/hip_runtime.h>
#include <hip/hip_bf16.h>

// ---------------------------------------------------------------------------
// Problem constants (from reference): B=4096, V=25, DIM=512, H=8, HD=64
// ---------------------------------------------------------------------------
#define B_SZ  4096
#define V_SZ  25
#define DIM_C 512
#define H_SZ  8
#define HD_C  64
#define MROWS (B_SZ * V_SZ)   // 102400 rows for the big GEMMs

typedef __attribute__((ext_vector_type(16))) __bf16 v16bf;
typedef __attribute__((ext_vector_type(8)))  __bf16 v8bf;
typedef __attribute__((ext_vector_type(8)))  float  v8f;
typedef __attribute__((ext_vector_type(4)))  int    v4i;

// CDNA5 async global->LDS direct copy (ASYNCcnt-tracked), when the toolchain
// exposes it; otherwise fall back to VGPR round-trip staging.
#if __has_builtin(__builtin_amdgcn_global_load_async_to_lds_b128)
#define GEMM_ASYNC 1
#else
#define GEMM_ASYNC 0
#endif

// Builtin signature (from hipcc diagnostics): (v4i as1* src, v4i as3* dst, Ii, Ii)
#define ASYNC_G(p) ((__attribute__((address_space(1))) v4i*)(p))
#define ASYNC_L(p) ((__attribute__((address_space(3))) v4i*)(p))

__device__ __forceinline__ void wait_asynccnt0() {
#if __has_builtin(__builtin_amdgcn_s_wait_asynccnt)
  __builtin_amdgcn_s_wait_asynccnt(0);
#else
  asm volatile("s_wait_asynccnt 0" ::: "memory");
#endif
}

// Build a 16-element bf16 WMMA operand from two contiguous 16-byte LDS reads.
__device__ __forceinline__ v16bf ld_frag16(const __bf16* p_lo, const __bf16* p_hi) {
  v8bf lo = *(const v8bf*)p_lo;
  v8bf hi = *(const v8bf*)p_hi;
  return __builtin_shufflevector(lo, hi, 0,1,2,3,4,5,6,7,8,9,10,11,12,13,14,15);
}

// ---------------------------------------------------------------------------
// f32 -> bf16 conversion (row-major copy, vectorized)
// ---------------------------------------------------------------------------
__global__ __launch_bounds__(256) void cvt_f32_to_bf16(const float* __restrict__ in,
                                                       __bf16* __restrict__ out, int n) {
  int i = (blockIdx.x * 256 + threadIdx.x) * 4;
  if (i + 3 < n) {
    float4 f = *(const float4*)(in + i);
    out[i + 0] = (__bf16)f.x;
    out[i + 1] = (__bf16)f.y;
    out[i + 2] = (__bf16)f.z;
    out[i + 3] = (__bf16)f.w;
  } else {
    for (; i < n; ++i) out[i] = (__bf16)in[i];
  }
}

// ---------------------------------------------------------------------------
// f32 [K,N] -> bf16 [N,K] transpose (weights only; tiny, one-time cost).
// Makes GEMM B-tile staging a contiguous copy => async-DMA friendly.
// ---------------------------------------------------------------------------
__global__ __launch_bounds__(256) void cvt_transpose_to_bf16(
    const float* __restrict__ in, __bf16* __restrict__ out, int K, int N) {
  __shared__ float t[32][33];
  const int kb = blockIdx.x * 32, nb = blockIdx.y * 32;
  const int tx = threadIdx.x & 31, ty = threadIdx.x >> 5;   // 32x8
  #pragma unroll
  for (int i = ty; i < 32; i += 8)
    t[i][tx] = in[(size_t)(kb + i) * N + nb + tx];
  __syncthreads();
  #pragma unroll
  for (int i = ty; i < 32; i += 8)
    out[(size_t)(nb + i) * K + kb + tx] = (__bf16)t[tx][i];
}

// ---------------------------------------------------------------------------
// bf16 WMMA GEMM: C[M,N] = A[M,K] * Bt[N,K]^T (+bias, f32 out, when FINAL)
// Block tile 128x128, BK=32, 8 waves (4x2), wave tile 32x64, double-buffered
// LDS. Both A ([m][k]) and B ([n][k], already transposed in global) stage as
// contiguous 16B chunks -> async global->LDS DMA when available.
// WMMA 16-bit operand layouts (CDNA5):
//   A (16x32): lane m=L%16; elems 0-7 -> K = half*8+i, elems 8-15 -> 16+half*8+i
//   B (32x16): lane n=L%16; elems 0-7 -> K = half*8+i, elems 8-15 -> 16+half*8+i
// ---------------------------------------------------------------------------
template<bool FINAL>
__global__ __launch_bounds__(256) void gemm_bf16_wmma(
    const __bf16* __restrict__ A, const __bf16* __restrict__ Bt,
    void* __restrict__ Cout, const float* __restrict__ bias,
    int M, int N, int K)
{
  constexpr int BM = 128, BN = 128, BK = 32, LDP = BK + 8;
  __shared__ __bf16 sA[2][BM][LDP];
  __shared__ __bf16 sB[2][BN][LDP];   // [n][k]

  const int tid  = threadIdx.x;
  const int wave = tid >> 5;
  const int lane = tid & 31;
  const int lm   = lane & 15;
  const int hf   = lane >> 4;
  const int waveM = (wave >> 1) * 32;
  const int waveN = (wave & 1) * 64;
  const int mBase = blockIdx.y * BM;
  const int nBase = blockIdx.x * BN;
  // 512 16B chunks per 8KB tile; chunk c: row = c>>2, elem-off = (c&3)*8
  const int c0 = tid, c1 = tid + 256;
  const int r0 = c0 >> 2, o0 = (c0 & 3) * 8;
  const int r1 = c1 >> 2, o1 = (c1 & 3) * 8;

#if GEMM_ASYNC
  auto stage = [&](int buf, int k0) {
    __builtin_amdgcn_global_load_async_to_lds_b128(
        ASYNC_G(A + (size_t)(mBase + r0) * K + k0 + o0), ASYNC_L(&sA[buf][r0][o0]), 0, 0);
    __builtin_amdgcn_global_load_async_to_lds_b128(
        ASYNC_G(A + (size_t)(mBase + r1) * K + k0 + o1), ASYNC_L(&sA[buf][r1][o1]), 0, 0);
    __builtin_amdgcn_global_load_async_to_lds_b128(
        ASYNC_G(Bt + (size_t)(nBase + r0) * K + k0 + o0), ASYNC_L(&sB[buf][r0][o0]), 0, 0);
    __builtin_amdgcn_global_load_async_to_lds_b128(
        ASYNC_G(Bt + (size_t)(nBase + r1) * K + k0 + o1), ASYNC_L(&sB[buf][r1][o1]), 0, 0);
  };
  auto stage_wait = [&]() { wait_asynccnt0(); };
#else
  auto stage = [&](int buf, int k0) {
    uint4 ra0 = *(const uint4*)(A + (size_t)(mBase + r0) * K + k0 + o0);
    uint4 ra1 = *(const uint4*)(A + (size_t)(mBase + r1) * K + k0 + o1);
    uint4 rb0 = *(const uint4*)(Bt + (size_t)(nBase + r0) * K + k0 + o0);
    uint4 rb1 = *(const uint4*)(Bt + (size_t)(nBase + r1) * K + k0 + o1);
    *(uint4*)&sA[buf][r0][o0] = ra0;
    *(uint4*)&sA[buf][r1][o1] = ra1;
    *(uint4*)&sB[buf][r0][o0] = rb0;
    *(uint4*)&sB[buf][r1][o1] = rb1;
  };
  auto stage_wait = [&]() {};
#endif

  v8f acc[2][4] = {};
  stage(0, 0);
  stage_wait();
  __syncthreads();

  const int steps = K / BK;
  int cur = 0;
  for (int kt = 0; kt < steps; ++kt) {
    if (kt + 1 < steps) stage(cur ^ 1, (kt + 1) * BK);   // overlaps with WMMAs

    v16bf afrag[2], bfrag[4];
    #pragma unroll
    for (int tm = 0; tm < 2; ++tm) {
      const __bf16* rp = &sA[cur][waveM + tm * 16 + lm][0];
      afrag[tm] = ld_frag16(rp + hf * 8, rp + 16 + hf * 8);
    }
    #pragma unroll
    for (int tn = 0; tn < 4; ++tn) {
      const __bf16* cp = &sB[cur][waveN + tn * 16 + lm][0];
      bfrag[tn] = ld_frag16(cp + hf * 8, cp + 16 + hf * 8);
    }
    #pragma unroll
    for (int tm = 0; tm < 2; ++tm)
      #pragma unroll
      for (int tn = 0; tn < 4; ++tn)
        acc[tm][tn] = __builtin_amdgcn_wmma_f32_16x16x32_bf16(
            false, afrag[tm], false, bfrag[tn], (short)0, acc[tm][tn], false, false);

    if (kt + 1 < steps) {
      stage_wait();
      __syncthreads();
      cur ^= 1;
    }
  }

  // C layout: element j of acc -> row = hf*8 + j, col = lane&15 (per tile)
  #pragma unroll
  for (int tm = 0; tm < 2; ++tm) {
    #pragma unroll
    for (int tn = 0; tn < 4; ++tn) {
      const int col = nBase + waveN + tn * 16 + lm;
      #pragma unroll
      for (int j = 0; j < 8; ++j) {
        const int row = mBase + waveM + tm * 16 + hf * 8 + j;
        float v = acc[tm][tn][j];
        if constexpr (FINAL) {
          ((float*)Cout)[(size_t)row * N + col] = v + bias[col];
        } else {
          ((__bf16*)Cout)[(size_t)row * N + col] = (__bf16)v;
        }
      }
    }
  }
}

// ---------------------------------------------------------------------------
// Attention: one wave per (b,h). V=25 padded to 32.
// Irregular reshape: element (h, w, d) of q/e_k/vv lives at flat
//   base(b,h) + d*V + w   within the per-batch [V*DIM] (resp. [V*2DIM]) block.
// Scores S = q @ e_k^T (32x32, K=64), softmax((S + d_term)*HD^-0.5),
// O = P @ V (32x64, K=32). Output written directly as Y[b, w, h*64 + d].
// ---------------------------------------------------------------------------
__global__ __launch_bounds__(32) void attn_wmma(
    const __bf16* __restrict__ Qb,    // [B, H*HD*V] = [B, 12800]
    const __bf16* __restrict__ KVb,   // [B, 2*H*HD*V] = [B, 25600]
    const float*  __restrict__ e,     // [B, V*DIM] = [B, 12800]
    const float*  __restrict__ w1,    // [H*HD]
    __bf16* __restrict__ Yb)          // [B*V, DIM]
{
  __shared__ __bf16 q_s[32][64];    // [w][d]
  __shared__ __bf16 e_s[32][64];    // [w][d]
  __shared__ __bf16 vt_s[64][32];   // transposed: [d][w]
  __shared__ __bf16 p_s[32][32];    // softmax probs [w][w']
  __shared__ float  s_s[32][32];    // raw scores
  __shared__ float  dt[32];         // d_term per key w'

  const int lane = threadIdx.x;
  const int lm = lane & 15, hf = lane >> 4;
  const int bh = blockIdx.x;
  const int b = bh >> 3, h = bh & 7;
  const size_t qbase  = (size_t)b * (H_SZ * HD_C * V_SZ) + (size_t)h * (HD_C * V_SZ);
  const size_t kvbase = (size_t)b * (2 * H_SZ * HD_C * V_SZ)
                      + (size_t)(H_SZ * HD_C * V_SZ) + (size_t)h * (HD_C * V_SZ);

  // Gather with zero-padding rows 25..31
  for (int idx = lane; idx < 32 * 64; idx += 32) {
    const int w = idx & 31, d = idx >> 5;
    __bf16 qv = (__bf16)0.f, ev = (__bf16)0.f, vv = (__bf16)0.f;
    if (w < V_SZ) {
      const size_t fo = (size_t)d * V_SZ + w;
      qv = Qb[qbase + fo];
      ev = (__bf16)e[qbase + fo];
      vv = KVb[kvbase + fo];
    }
    q_s[w][d] = qv; e_s[w][d] = ev; vt_s[d][w] = vv;
  }
  __syncthreads();

  // d_term[w'] = sum_d w1[h,d] * e_k[w',d]
  {
    float a = 0.f;
    if (lane < V_SZ) {
      #pragma unroll
      for (int d = 0; d < HD_C; ++d) a += w1[h * HD_C + d] * (float)e_s[lane][d];
    }
    dt[lane] = a;
  }
  __syncthreads();

  // Scores: [32x32] = q[32x64] @ e_k^T[64x32], two K=32 steps
  v8f sc[2][2] = {};
  #pragma unroll
  for (int ks = 0; ks < 2; ++ks) {
    v16bf afrag[2], bfrag[2];
    #pragma unroll
    for (int tm = 0; tm < 2; ++tm) {
      const __bf16* rp = &q_s[tm * 16 + lm][ks * 32];
      afrag[tm] = ld_frag16(rp + hf * 8, rp + 16 + hf * 8);
    }
    #pragma unroll
    for (int tn = 0; tn < 2; ++tn) {
      const __bf16* cp = &e_s[tn * 16 + lm][ks * 32];  // B[k][n] = e_s[n][k]
      bfrag[tn] = ld_frag16(cp + hf * 8, cp + 16 + hf * 8);
    }
    #pragma unroll
    for (int tm = 0; tm < 2; ++tm)
      #pragma unroll
      for (int tn = 0; tn < 2; ++tn)
        sc[tm][tn] = __builtin_amdgcn_wmma_f32_16x16x32_bf16(
            false, afrag[tm], false, bfrag[tn], (short)0, sc[tm][tn], false, false);
  }
  #pragma unroll
  for (int tm = 0; tm < 2; ++tm)
    #pragma unroll
    for (int tn = 0; tn < 2; ++tn)
      #pragma unroll
      for (int j = 0; j < 8; ++j)
        s_s[tm * 16 + hf * 8 + j][tn * 16 + lm] = sc[tm][tn][j];
  __syncthreads();

  // Softmax over keys (25 valid), bias dt on key axis, scale = HD^-0.5
  {
    const int r = lane;
    const float scale = 0.125f;
    float v[V_SZ];
    float mx = -3.0e38f;
    #pragma unroll
    for (int j = 0; j < V_SZ; ++j) {
      v[j] = (s_s[r][j] + dt[j]) * scale;
      mx = fmaxf(mx, v[j]);
    }
    float sum = 0.f;
    #pragma unroll
    for (int j = 0; j < V_SZ; ++j) { v[j] = __expf(v[j] - mx); sum += v[j]; }
    const float inv = 1.f / sum;
    #pragma unroll
    for (int j = 0; j < V_SZ; ++j) p_s[r][j] = (__bf16)(v[j] * inv);
    #pragma unroll
    for (int j = V_SZ; j < 32; ++j) p_s[r][j] = (__bf16)0.f;
  }
  __syncthreads();

  // O[32x64] = P[32x32] @ V[32x64], single K=32 step
  v8f oc[2][4] = {};
  {
    v16bf afrag[2], bfrag[4];
    #pragma unroll
    for (int tm = 0; tm < 2; ++tm) {
      const __bf16* rp = &p_s[tm * 16 + lm][0];
      afrag[tm] = ld_frag16(rp + hf * 8, rp + 16 + hf * 8);
    }
    #pragma unroll
    for (int tn = 0; tn < 4; ++tn) {
      const __bf16* cp = &vt_s[tn * 16 + lm][0];     // B[k][n] = vt_s[n][k]
      bfrag[tn] = ld_frag16(cp + hf * 8, cp + 16 + hf * 8);
    }
    #pragma unroll
    for (int tm = 0; tm < 2; ++tm)
      #pragma unroll
      for (int tn = 0; tn < 4; ++tn)
        oc[tm][tn] = __builtin_amdgcn_wmma_f32_16x16x32_bf16(
            false, afrag[tm], false, bfrag[tn], (short)0, oc[tm][tn], false, false);
  }
  // Y[b, w, h*64 + d]  (this IS the reference's final transpose chain)
  #pragma unroll
  for (int tm = 0; tm < 2; ++tm)
    #pragma unroll
    for (int tn = 0; tn < 4; ++tn)
      #pragma unroll
      for (int j = 0; j < 8; ++j) {
        const int m = tm * 16 + hf * 8 + j;
        if (m < V_SZ) {
          const int n = tn * 16 + lm;
          Yb[((size_t)b * V_SZ + m) * DIM_C + h * HD_C + n] = (__bf16)oc[tm][tn][j];
        }
      }
}

// ---------------------------------------------------------------------------
// Host launcher
// ---------------------------------------------------------------------------
extern "C" void kernel_launch(void* const* d_in, const int* in_sizes, int n_in,
                              void* d_out, int out_size, void* d_ws, size_t ws_size,
                              hipStream_t stream) {
  (void)in_sizes; (void)n_in; (void)out_size; (void)ws_size;
  const float* x     = (const float*)d_in[0];
  const float* e     = (const float*)d_in[1];
  const float* Wq    = (const float*)d_in[2];
  const float* Wkv   = (const float*)d_in[3];
  const float* w1    = (const float*)d_in[4];
  const float* Wproj = (const float*)d_in[5];
  const float* bproj = (const float*)d_in[6];

  size_t off = 0;
  auto carve = [&](size_t bytes) -> void* {
    void* p = (char*)d_ws + off;
    off += (bytes + 255) & ~(size_t)255;
    return p;
  };
  __bf16* Xb   = (__bf16*)carve((size_t)MROWS * DIM_C * 2);
  __bf16* Wqt  = (__bf16*)carve((size_t)DIM_C * DIM_C * 2);       // [N,K]
  __bf16* Wkvt = (__bf16*)carve((size_t)DIM_C * 2 * DIM_C * 2);   // [2N,K]
  __bf16* Wpt  = (__bf16*)carve((size_t)DIM_C * DIM_C * 2);       // [N,K]
  __bf16* Qb   = (__bf16*)carve((size_t)MROWS * DIM_C * 2);
  __bf16* KVb  = (__bf16*)carve((size_t)MROWS * 2 * DIM_C * 2);
  __bf16* Yb   = Xb;   // Xb dead after the two input GEMMs -> reuse for attn out

  // x -> bf16 row-major (GEMM A operand)
  {
    int n = MROWS * DIM_C;
    cvt_f32_to_bf16<<<(n / 4 + 255) / 256, 256, 0, stream>>>(x, Xb, n);
  }
  // weights -> bf16 transposed [N,K] (GEMM B operand, contiguous-K staging)
  cvt_transpose_to_bf16<<<dim3(DIM_C / 32, DIM_C / 32), 256, 0, stream>>>(Wq, Wqt, DIM_C, DIM_C);
  cvt_transpose_to_bf16<<<dim3(DIM_C / 32, 2 * DIM_C / 32), 256, 0, stream>>>(Wkv, Wkvt, DIM_C, 2 * DIM_C);
  cvt_transpose_to_bf16<<<dim3(DIM_C / 32, DIM_C / 32), 256, 0, stream>>>(Wproj, Wpt, DIM_C, DIM_C);

  dim3 blk(256);
  // Q = x @ Wq   (bf16 out)
  gemm_bf16_wmma<false><<<dim3(DIM_C / 128, MROWS / 128), blk, 0, stream>>>(
      Xb, Wqt, Qb, nullptr, MROWS, DIM_C, DIM_C);
  // KV = x @ Wkv (bf16 out)
  gemm_bf16_wmma<false><<<dim3(2 * DIM_C / 128, MROWS / 128), blk, 0, stream>>>(
      Xb, Wkvt, KVb, nullptr, MROWS, 2 * DIM_C, DIM_C);
  // Attention per (b,h)
  attn_wmma<<<B_SZ * H_SZ, 32, 0, stream>>>(Qb, KVb, e, w1, Yb);
  // out = Y @ Wproj + bproj (f32 out)
  gemm_bf16_wmma<true><<<dim3(DIM_C / 128, MROWS / 128), blk, 0, stream>>>(
      Yb, Wpt, d_out, bproj, MROWS, DIM_C, DIM_C);
}
